// SwinTransformerBlock_5403068858609
// MI455X (gfx1250) — matrix operand order repaired and checked by reading, not compiled, not run on hardware
//
#include <hip/hip_runtime.h>
#include <hip/hip_bf16.h>

typedef __attribute__((ext_vector_type(16))) _Float16 v16h;
typedef __attribute__((ext_vector_type(8)))  _Float16 h8;
typedef __attribute__((ext_vector_type(8)))  float    v8f;

#define TOKENS   131072      // 32 * 4096
#define CDIM     256
#define NWIN     2048        // 32 * 64 windows
#define LRELU_G  1.4142135623730951f
#define WROW     40          // padded LDS row stride (f16) for 32-wide k-tile

__device__ __forceinline__ v8f wmma_f16(v16h a, v16h b, v8f c) {
    return __builtin_amdgcn_wmma_f32_16x16x32_f16(false, a, false, b, (short)0, c,
                                                  false, false);
}

// windowed token t -> natural row (b*4096 + h*64 + w).  Same map is used for the
// input gather (shift -SHIFT + window partition) and output scatter (reverse).
__device__ __forceinline__ int permute_row(int t) {
    int wi = t >> 6, n = t & 63;
    int b  = wi >> 6, w = wi & 63;
    int hp = ((w >> 3) << 3) + (n >> 3);     // shifted-space row
    int wp = ((w & 7) << 3) + (n & 7);       // shifted-space col
    int h  = (hp + 4) & 63;
    int ww = (wp + 4) & 63;
    return (b << 12) | (h << 6) | ww;
}

__device__ __forceinline__ int region(int p) { return (p < 56) ? 0 : ((p < 60) ? 1 : 2); }

// A fragment: row-major f16 matrix, 16 rows starting at row0, K-tile at kc.
__device__ __forceinline__ v16h load_frag_a(const _Float16* p, int ld, int row0,
                                            int kc, int lane) {
    const _Float16* q = p + (size_t)(row0 + (lane & 15)) * ld + kc + ((lane & 16) ? 16 : 0);
    h8 lo = *(const h8*)q;
    h8 hi = *(const h8*)(q + 8);
    return __builtin_shufflevector(lo, hi, 0,1,2,3,4,5,6,7,8,9,10,11,12,13,14,15);
}

// B fragment: 16 consecutive f16 (one weight row segment)
__device__ __forceinline__ v16h load_frag_b16(const _Float16* p) {
    h8 lo = *(const h8*)p;
    h8 hi = *(const h8*)(p + 8);
    return __builtin_shufflevector(lo, hi, 0,1,2,3,4,5,6,7,8,9,10,11,12,13,14,15);
}

// ---------------------------------------------------------------------------
// Weight convert: f32 -> f16 with StyleGAN gain folded in.
__global__ void cvt_kernel(const float* __restrict__ src, _Float16* __restrict__ dst,
                           int n, float scale) {
    int i = blockIdx.x * blockDim.x + threadIdx.x;
    if (i < n) dst[i] = (_Float16)(src[i] * scale);
}

// ---------------------------------------------------------------------------
// Prep: gather windowed tokens, L2-normalize, emit xw_f16, normx_f16, x_f16(nat).
__global__ __launch_bounds__(256) void prep_kernel(const float* __restrict__ x,
                                                   _Float16* __restrict__ xw,
                                                   _Float16* __restrict__ nx,
                                                   _Float16* __restrict__ xnat) {
    int lane = threadIdx.x & 31, wave = threadIdx.x >> 5;
    int t   = blockIdx.x * 8 + wave;
    int src = permute_row(t);
    const float* xp = x + (size_t)src * CDIM + lane * 8;
    float v[8]; float ss = 0.f;
#pragma unroll
    for (int i = 0; i < 8; i++) { v[i] = xp[i]; ss += v[i] * v[i]; }
#pragma unroll
    for (int m = 1; m < 32; m <<= 1) ss += __shfl_xor(ss, m, 32);
    float inv = 1.f / fmaxf(sqrtf(ss), 1e-12f);
    size_t ot = (size_t)t * CDIM + lane * 8;
    size_t on = (size_t)src * CDIM + lane * 8;
#pragma unroll
    for (int i = 0; i < 8; i++) {
        _Float16 h = (_Float16)v[i];
        xw[ot + i]   = h;
        xnat[on + i] = h;
        nx[ot + i]   = (_Float16)(v[i] * inv);
    }
}

// ---------------------------------------------------------------------------
// Async-stage one 32(k) x 256(o) weight tile into LDS (padded rows of WROW f16).
// 1024 16-byte chunks, 4 per thread, via ASYNCcnt-tracked LDS DMA.
__device__ __forceinline__ void stage_w(const _Float16* W, int K, int o0, int k0,
                                        _Float16* dstbase, int tid) {
#pragma unroll
    for (int r = 0; r < 4; r++) {
        int c  = tid + r * 256;          // chunk id 0..1023
        int o  = c >> 2;                 // output row within chunk (0..255)
        int ch = c & 3;                  // 8-element group within k-tile
        unsigned lds = (unsigned)(uintptr_t)(dstbase + o * WROW + ch * 8);
        unsigned long long ga =
            (unsigned long long)(uintptr_t)(W + (size_t)(o0 + o) * K + k0 + ch * 8);
        asm volatile("global_load_async_to_lds_b128 %0, %1, off"
                     :: "v"(lds), "v"(ga) : "memory");
    }
}

// ---------------------------------------------------------------------------
// Generic WMMA GEMM: Y[t,o] = act( sum_k A[t,k] * W[o,k] + bias[o] )
// One wave per 16-row tile; 16 output tiles (256 cols) per wave; weight k-tiles
// double-buffered in LDS via async LDS-DMA; B fragments software-pipelined.
// flags: 1 = lrelu*sqrt(2), 2 = scatter rows through permute_row.
__global__ __launch_bounds__(256) void gemm_kernel(
    const _Float16* __restrict__ A, int lda,
    const _Float16* __restrict__ A2, int lda2, int ksplit,
    const _Float16* __restrict__ W,
    const float* __restrict__ bias,
    _Float16* __restrict__ Yh, float* __restrict__ Yf,
    int T, int K, int O, int flags) {
    __shared__ _Float16 sw[2][256 * WROW];          // 2 x 20 KB
    const int tid  = threadIdx.x;
    const int lane = tid & 31;
    const int wave = tid >> 5;
    const int row0 = (blockIdx.x * 8 + wave) * 16;  // grid covers T exactly
    const int o0   = blockIdx.y * 256;
    const int l15  = lane & 15;
    const int hsel = (lane & 16) ? 8 : 0;
    const int bsel = (lane & 16) ? 16 : 0;

    v8f acc[16];
    const v8f z = {0.f,0.f,0.f,0.f,0.f,0.f,0.f,0.f};
#pragma unroll
    for (int i = 0; i < 16; i++) acc[i] = z;

    const int nk = K >> 5;
    stage_w(W, K, o0, 0, &sw[0][0], tid);           // prologue: tile 0

    for (int ki = 0; ki < nk; ki++) {
        const int k0 = ki << 5;
        asm volatile("s_wait_asynccnt 0x0" ::: "memory");
        __syncthreads();                             // tile ki visible to all waves
        if (ki + 1 < nk)
            stage_w(W, K, o0, k0 + 32, &sw[(ki + 1) & 1][0], tid);

        const _Float16* Ap = A; int kc = k0; int ld = lda;
        if (A2 != nullptr && k0 >= ksplit) { Ap = A2; kc = k0 - ksplit; ld = lda2; }
        if (ki + 1 < nk)                             // hint next A fragment
            __builtin_prefetch(Ap + (size_t)(row0 + l15) * ld + kc + 32, 0, 3);
        v16h a = load_frag_a(Ap, ld, row0, kc, lane);

        const _Float16* wl = &sw[ki & 1][0] + (size_t)l15 * WROW + bsel;
        v16h bcur = load_frag_b16(wl);               // pipelined B fragments
#pragma unroll
        for (int ot = 0; ot < 16; ot++) {
            v16h bnext = bcur;
            if (ot + 1 < 16)
                bnext = load_frag_b16(wl + (size_t)(ot + 1) * 16 * WROW);
            acc[ot] = wmma_f16(a, bcur, acc[ot]);
            bcur = bnext;
        }
    }

#pragma unroll
    for (int ot = 0; ot < 16; ot++) {
        int col = o0 + ot * 16 + l15;
        float bv = bias ? bias[col] : 0.f;
#pragma unroll
        for (int j = 0; j < 8; j++) {
            int row = row0 + j + hsel;
            float v = acc[ot][j] + bv;
            if (flags & 1) v = (v > 0.f ? v : 0.2f * v) * LRELU_G;
            int r = (flags & 2) ? permute_row(row) : row;
            size_t idx = (size_t)r * O + col;
            if (Yh) Yh[idx] = (_Float16)v;
            if (Yf) Yf[idx] = v;
        }
    }
}

// ---------------------------------------------------------------------------
// Windowed attention: one block per window, one wave per head.
// S = Q·K^T * scale + mask ; P = softmax(S) ; O = P·V
__global__ __launch_bounds__(256) void attn_kernel(const _Float16* __restrict__ Q,
                                                   const _Float16* __restrict__ Km,
                                                   const _Float16* __restrict__ V,
                                                   _Float16* __restrict__ Out) {
    __shared__ _Float16 Pl[8 * 64 * 64];   // per-wave 64x64 f16 P tile (64 KB)
    const int lane = threadIdx.x & 31;
    const int head = threadIdx.x >> 5;
    const int win  = blockIdx.x;
    const int wimg = win & 63;
    const int wh = wimg >> 3, ww = wimg & 7;
    const int l15  = lane & 15;
    const int half8 = (lane & 16) ? 8 : 0;
    const int bsel  = (lane & 16) ? 16 : 0;
    const size_t base = (size_t)win * 64 * CDIM + head * 32;
    const float scale = 0.17677669529663687f;  // 1/sqrt(32)
    _Float16* Pw = Pl + head * 4096;
    const v8f z = {0.f,0.f,0.f,0.f,0.f,0.f,0.f,0.f};

    // B fragments of K (rows of K == columns of K^T), reused across all m-tiles
    v16h bk[4];
#pragma unroll
    for (int nt = 0; nt < 4; nt++)
        bk[nt] = *(const v16h*)(Km + base + (size_t)(nt * 16 + l15) * CDIM + bsel);

    // column region ids (key side) for the mask
    int cc[4];
#pragma unroll
    for (int nt = 0; nt < 4; nt++) {
        int n = nt * 16 + l15;
        cc[nt] = region(wh * 8 + (n >> 3)) * 3 + region(ww * 8 + (n & 7));
    }

    for (int mt = 0; mt < 4; mt++) {
        v16h a = load_frag_a(Q + base, CDIM, mt * 16, 0, lane);
        v8f s[4];
#pragma unroll
        for (int nt = 0; nt < 4; nt++) s[nt] = wmma_f16(a, bk[nt], z);

#pragma unroll
        for (int j = 0; j < 8; j++) {
            int nr = mt * 16 + j + half8;
            int cr = region(wh * 8 + (nr >> 3)) * 3 + region(ww * 8 + (nr & 7));
            float vals[4]; float vmax = -1e30f;
#pragma unroll
            for (int nt = 0; nt < 4; nt++) {
                float vv = s[nt][j] * scale + ((cc[nt] != cr) ? -100.f : 0.f);
                vals[nt] = vv; vmax = fmaxf(vmax, vv);
            }
#pragma unroll
            for (int m = 1; m < 16; m <<= 1) vmax = fmaxf(vmax, __shfl_xor(vmax, m, 16));
            float e[4]; float sum = 0.f;
#pragma unroll
            for (int nt = 0; nt < 4; nt++) { e[nt] = __expf(vals[nt] - vmax); sum += e[nt]; }
#pragma unroll
            for (int m = 1; m < 16; m <<= 1) sum += __shfl_xor(sum, m, 16);
            float inv = 1.f / sum;
#pragma unroll
            for (int nt = 0; nt < 4; nt++)
                Pw[nr * 64 + nt * 16 + l15] = (_Float16)(e[nt] * inv);
        }
    }
    __syncthreads();

    // B fragments of V (column-major gather: lane = output channel)
    v16h bv[2][2];
#pragma unroll
    for (int ks = 0; ks < 2; ks++)
#pragma unroll
        for (int nt = 0; nt < 2; nt++)
#pragma unroll
            for (int e = 0; e < 16; e++)
                bv[ks][nt][e] =
                    V[base + (size_t)(ks * 32 + bsel + e) * CDIM + nt * 16 + l15];

    for (int mt = 0; mt < 4; mt++) {
        v8f o0 = z, o1 = z;
#pragma unroll
        for (int ks = 0; ks < 2; ks++) {
            v16h a = load_frag_a(Pw, 64, mt * 16, ks * 32, lane);
            o0 = wmma_f16(a, bv[ks][0], o0);
            o1 = wmma_f16(a, bv[ks][1], o1);
        }
#pragma unroll
        for (int j = 0; j < 8; j++) {
            size_t row = (size_t)win * 64 + mt * 16 + j + half8;
            Out[row * CDIM + head * 32 + l15]      = (_Float16)o0[j];
            Out[row * CDIM + head * 32 + 16 + l15] = (_Float16)o1[j];
        }
    }
}

// ---------------------------------------------------------------------------
extern "C" void kernel_launch(void* const* d_in, const int* in_sizes, int n_in,
                              void* d_out, int out_size, void* d_ws, size_t ws_size,
                              hipStream_t stream) {
    const float* x      = (const float*)d_in[0];
    const float* q_w    = (const float*)d_in[1];
    const float* q_b    = (const float*)d_in[2];
    const float* k_w    = (const float*)d_in[3];
    const float* k_b    = (const float*)d_in[4];
    const float* v_w    = (const float*)d_in[5];
    const float* v_b    = (const float*)d_in[6];
    const float* proj_w = (const float*)d_in[7];
    const float* proj_b = (const float*)d_in[8];
    const float* fuse_w = (const float*)d_in[9];
    const float* fuse_b = (const float*)d_in[10];
    const float* fc1_w  = (const float*)d_in[11];
    const float* fc1_b  = (const float*)d_in[12];
    const float* fc2_w  = (const float*)d_in[13];
    const float* fc2_b  = (const float*)d_in[14];

    char* ws = (char*)d_ws;
    const size_t BUF = (size_t)TOKENS * CDIM * sizeof(_Float16);  // 64 MB
    _Float16* b0 = (_Float16*)(ws + 0 * BUF);   // xw (windowed) -> proj out (nat)
    _Float16* b1 = (_Float16*)(ws + 1 * BUF);   // normx -> attn out (windowed)
    _Float16* b2 = (_Float16*)(ws + 2 * BUF);   // q -> fuse out
    _Float16* b3 = (_Float16*)(ws + 3 * BUF);   // k
    _Float16* b4 = (_Float16*)(ws + 4 * BUF);   // v
    _Float16* b5 = (_Float16*)(ws + 5 * BUF);   // x f16 (natural order)
    _Float16* bh = (_Float16*)(ws + 6 * BUF);   // fc1 out: 131072x1024 (256 MB)
    char* wbase = ws + 6 * BUF + (size_t)TOKENS * 1024 * sizeof(_Float16);
    _Float16* wq  = (_Float16*)wbase;
    _Float16* wk  = wq  + 65536;
    _Float16* wv  = wk  + 65536;
    _Float16* wpj = wv  + 65536;
    _Float16* wf  = wpj + 65536;     // 256 x 512
    _Float16* w1  = wf  + 131072;    // 1024 x 256
    _Float16* w2  = w1  + 262144;    // 256 x 1024

    // weight conversion (gain = 1/sqrt(in_features))
    cvt_kernel<<<256, 256, 0, stream>>>(q_w,   wq,  65536,  0.0625f);
    cvt_kernel<<<256, 256, 0, stream>>>(k_w,   wk,  65536,  0.0625f);
    cvt_kernel<<<256, 256, 0, stream>>>(v_w,   wv,  65536,  0.0625f);
    cvt_kernel<<<256, 256, 0, stream>>>(proj_w,wpj, 65536,  0.0625f);
    cvt_kernel<<<512, 256, 0, stream>>>(fuse_w,wf,  131072, 0.04419417382415922f);
    cvt_kernel<<<1024,256, 0, stream>>>(fc1_w, w1,  262144, 0.0625f);
    cvt_kernel<<<1024,256, 0, stream>>>(fc2_w, w2,  262144, 0.03125f);

    prep_kernel<<<TOKENS / 8, 256, 0, stream>>>(x, b0, b1, b5);

    dim3 g1(TOKENS / 16 / 8, 1);
    // q, k (from normalized x), v (from raw windowed x)
    gemm_kernel<<<g1, 256, 0, stream>>>(b1, CDIM, nullptr, 0, 0, wq,  q_b,   b2, nullptr, TOKENS, 256, 256, 0);
    gemm_kernel<<<g1, 256, 0, stream>>>(b1, CDIM, nullptr, 0, 0, wk,  k_b,   b3, nullptr, TOKENS, 256, 256, 0);
    gemm_kernel<<<g1, 256, 0, stream>>>(b0, CDIM, nullptr, 0, 0, wv,  v_b,   b4, nullptr, TOKENS, 256, 256, 0);

    attn_kernel<<<NWIN, 256, 0, stream>>>(b2, b3, b4, b1);

    // proj + window-reverse + un-shift (scatter)
    gemm_kernel<<<g1, 256, 0, stream>>>(b1, CDIM, nullptr, 0, 0, wpj, proj_b, b0, nullptr, TOKENS, 256, 256, 2);
    // fuse: concat[shortcut, out] @ fuse_w^T, lrelu
    gemm_kernel<<<g1, 256, 0, stream>>>(b5, CDIM, b0, CDIM, 256, wf, fuse_b,  b2, nullptr, TOKENS, 512, 256, 1);
    // fc1: 256 -> 1024, lrelu
    dim3 g2(TOKENS / 16 / 8, 4);
    gemm_kernel<<<g2, 256, 0, stream>>>(b2, CDIM, nullptr, 0, 0, w1,  fc1_b,  bh, nullptr, TOKENS, 256, 1024, 1);
    // fc2: 1024 -> 256, f32 output
    gemm_kernel<<<g1, 256, 0, stream>>>(bh, 1024, nullptr, 0, 0, w2,  fc2_b,  nullptr, (float*)d_out, TOKENS, 1024, 256, 0);

    (void)in_sizes; (void)n_in; (void)out_size; (void)ws_size;
}
